// GraphUNet_62749472195033
// MI455X (gfx1250) — matrix operand order repaired and checked by reading, not compiled, not run on hardware
//
#include <hip/hip_runtime.h>
#include <hip/hip_bf16.h>

typedef __attribute__((ext_vector_type(16))) __bf16 v16bf;
typedef __attribute__((ext_vector_type(8)))  float  v8f;

#define DD 64

// ---------------------------------------------------------------------------
// Fused 2-layer MLP on 16-row tiles per wave using v_wmma_f32_16x16x32_bf16.
//   out[R x 64] = relu(A[R x K1] @ W1[K1 x 64] + b1) @ W2[64 x 64] + b2 (+resid)
// A comes from in0 (stride K1) or, when CONCAT, from [in0 | in1] (stride 64+64).
// ---------------------------------------------------------------------------
template<int K1, bool CONCAT>
__global__ __launch_bounds__(128) void mlp2_wmma_kernel(
    const float* __restrict__ in0, const float* __restrict__ in1,
    const float* __restrict__ W1,  const float* __restrict__ b1,
    const float* __restrict__ W2,  const float* __restrict__ b2,
    const float* __restrict__ residual,
    float* __restrict__ out, int R)
{
    constexpr int KP      = (K1 + 31) & ~31;        // K1 padded to mult of 32
    constexpr int KSTEPS1 = KP / 32;
    constexpr int SSTR    = (KP > 64) ? KP : 64;    // stage row stride (elems)

    __shared__ __align__(16) __bf16 sW1t[DD * KP];  // transposed: [n][k]
    __shared__ __align__(16) __bf16 sW2t[DD * DD];  // transposed: [n][k]
    __shared__ __align__(16) __bf16 stage[4 * 16 * SSTR];

    const int tid  = threadIdx.x;
    const int lane = tid & 31;
    const int wave = tid >> 5;
    const int col  = lane & 15;      // row for A-frag, col for B/C-frag
    const int hi   = lane >> 4;      // 0 or 1 (lane half)
    __bf16* st = stage + wave * 16 * SSTR;

    // --- stage transposed bf16 weights (zero-pad W1 rows K1..KP) ---
    for (int idx = tid; idx < DD * KP; idx += 128) {
        int n = idx / KP, k = idx % KP;
        sW1t[idx] = (k < K1) ? (__bf16)W1[k * DD + n] : (__bf16)0.f;
    }
    for (int idx = tid; idx < DD * DD; idx += 128) {
        int n = idx / DD, k = idx % DD;
        sW2t[idx] = (__bf16)W2[k * DD + n];
    }

    // --- stage this wave's 16 input rows as bf16 (zero-pad OOB/K-pad) ---
    const int rowBase = blockIdx.x * 64 + wave * 16;
    for (int e = lane; e < 16 * KP; e += 32) {
        int row = e / KP, k = e % KP;
        int gr = rowBase + row;
        float v = 0.f;
        if (gr < R && k < K1) {
            if (CONCAT) v = (k < DD) ? in0[(size_t)gr * DD + k]
                                     : in1[(size_t)gr * DD + (k - DD)];
            else        v = in0[(size_t)gr * K1 + k];
        }
        st[row * SSTR + k] = (__bf16)v;
    }
    __syncthreads();

    // ---- matmul 1: bias init, accumulate KSTEPS1 WMMAs per 16-col tile ----
    v8f acc[4];
#pragma unroll
    for (int nt = 0; nt < 4; ++nt) {
        float bv = b1[nt * 16 + col];
#pragma unroll
        for (int i = 0; i < 8; ++i) acc[nt][i] = bv;
    }
#pragma unroll
    for (int ks = 0; ks < KSTEPS1; ++ks) {
        v16bf a;  // A frag: lane holds row=col, K groups {hi*8.., 16+hi*8..}
        {
            const __bf16* p0 = st + (size_t)col * SSTR + ks * 32 + hi * 8;
            ((float4*)&a)[0] = *(const float4*)p0;
            ((float4*)&a)[1] = *(const float4*)(p0 + 16);
        }
#pragma unroll
        for (int nt = 0; nt < 4; ++nt) {
            v16bf b;  // B frag: lane holds col, K groups {hi*8.., 16+hi*8..} (transposed W)
            const __bf16* pb = sW1t + (size_t)(nt * 16 + col) * KP + ks * 32 + hi * 8;
            ((float4*)&b)[0] = ((const float4*)pb)[0];
            ((float4*)&b)[1] = ((const float4*)pb)[2];
            acc[nt] = __builtin_amdgcn_wmma_f32_16x16x32_bf16(
                false, a, false, b, (short)0, acc[nt], false, false);
        }
    }
    __syncthreads();

    // relu + restage the 16x64 intermediate as bf16 (C layout -> row major)
#pragma unroll
    for (int nt = 0; nt < 4; ++nt)
#pragma unroll
        for (int v = 0; v < 8; ++v) {
            int r0 = v + hi * 8;
            st[r0 * SSTR + nt * 16 + col] = (__bf16)fmaxf(acc[nt][v], 0.f);
        }
    __syncthreads();

    // ---- matmul 2: [16 x 64] @ [64 x 64] + b2 (+residual) ----
    v8f acc2[4];
#pragma unroll
    for (int nt = 0; nt < 4; ++nt) {
        float bv = b2[nt * 16 + col];
#pragma unroll
        for (int i = 0; i < 8; ++i) acc2[nt][i] = bv;
    }
#pragma unroll
    for (int ks = 0; ks < 2; ++ks) {
        v16bf a;
        {
            const __bf16* p0 = st + (size_t)col * SSTR + ks * 32 + hi * 8;
            ((float4*)&a)[0] = *(const float4*)p0;
            ((float4*)&a)[1] = *(const float4*)(p0 + 16);
        }
#pragma unroll
        for (int nt = 0; nt < 4; ++nt) {
            v16bf b;
            const __bf16* pb = sW2t + (size_t)(nt * 16 + col) * DD + ks * 32 + hi * 8;
            ((float4*)&b)[0] = ((const float4*)pb)[0];
            ((float4*)&b)[1] = ((const float4*)pb)[2];
            acc2[nt] = __builtin_amdgcn_wmma_f32_16x16x32_bf16(
                false, a, false, b, (short)0, acc2[nt], false, false);
        }
    }

    // epilogue: optional residual, store f32
#pragma unroll
    for (int nt = 0; nt < 4; ++nt)
#pragma unroll
        for (int v = 0; v < 8; ++v) {
            int gr = rowBase + v + hi * 8;
            if (gr < R) {
                size_t o = (size_t)gr * DD + nt * 16 + col;
                float val = acc2[nt][v];
                if (residual) val += residual[o];
                out[o] = val;
            }
        }
}

// aggr[r[e]][:] += msg[s[e]][:]  -- 16 threads per edge, float4 gather + atomics
__global__ void scatter_add_kernel(const float* __restrict__ msg,
                                   const int* __restrict__ s,
                                   const int* __restrict__ r,
                                   float* __restrict__ aggr, int E)
{
    long long t = (long long)blockIdx.x * blockDim.x + threadIdx.x;
    int e = (int)(t >> 4);
    if (e >= E) return;
    int c4 = ((int)t & 15) * 4;
    float4 m = *(const float4*)(msg + (size_t)s[e] * DD + c4);
    float* dst = aggr + (size_t)r[e] * DD + c4;
    atomicAdd(dst + 0, m.x);
    atomicAdd(dst + 1, m.y);
    atomicAdd(dst + 2, m.z);
    atomicAdd(dst + 3, m.w);
}

// dst[row][:] = src[idx[row]][:]
__global__ void gather_kernel(const float* __restrict__ src,
                              const int* __restrict__ idx,
                              float* __restrict__ dst, int R)
{
    long long t = (long long)blockIdx.x * blockDim.x + threadIdx.x;
    int row = (int)(t >> 4);
    if (row >= R) return;
    int c4 = ((int)t & 15) * 4;
    *(float4*)(dst + (size_t)row * DD + c4) =
        *(const float4*)(src + (size_t)idx[row] * DD + c4);
}

__global__ void zero_kernel(float* __restrict__ p, long long n)
{
    long long i = (long long)blockIdx.x * blockDim.x + threadIdx.x;
    if (i < n) p[i] = 0.f;
}

// out[N x 2] = relu(h @ Wd1 + bd1) @ Wd2 + bd2   (tiny: scalar VALU)
__global__ __launch_bounds__(128) void decode_kernel(
    const float* __restrict__ h,
    const float* __restrict__ Wd1, const float* __restrict__ bd1,
    const float* __restrict__ Wd2, const float* __restrict__ bd2,
    float* __restrict__ out, int N)
{
    __shared__ float sW1[64 * 64];
    __shared__ float sW2[128];
    for (int i = threadIdx.x; i < 64 * 64; i += 128) sW1[i] = Wd1[i];
    for (int i = threadIdx.x; i < 128; i += 128) sW2[i] = Wd2[i];
    __syncthreads();
    int i = blockIdx.x * 128 + threadIdx.x;
    if (i >= N) return;
    float hr[64];
#pragma unroll
    for (int k4 = 0; k4 < 16; ++k4)
        *(float4*)(hr + 4 * k4) = *(const float4*)(h + (size_t)i * 64 + 4 * k4);
    float o0 = bd2[0], o1 = bd2[1];
    for (int j = 0; j < 64; ++j) {
        float acc = bd1[j];
#pragma unroll
        for (int k = 0; k < 64; ++k) acc = fmaf(hr[k], sW1[k * 64 + j], acc);
        acc = fmaxf(acc, 0.f);
        o0 = fmaf(acc, sW2[2 * j], o0);
        o1 = fmaf(acc, sW2[2 * j + 1], o1);
    }
    out[(size_t)i * 2]     = o0;
    out[(size_t)i * 2 + 1] = o1;
}

extern "C" void kernel_launch(void* const* d_in, const int* in_sizes, int n_in,
                              void* d_out, int out_size, void* d_ws, size_t ws_size,
                              hipStream_t stream)
{
    const float* x    = (const float*)d_in[0];
    const float* We1  = (const float*)d_in[1];
    const float* be1  = (const float*)d_in[2];
    const float* We2  = (const float*)d_in[3];
    const float* be2  = (const float*)d_in[4];
    const float* Wm1  = (const float*)d_in[5];
    const float* bm1  = (const float*)d_in[6];
    const float* Wm2  = (const float*)d_in[7];
    const float* bm2  = (const float*)d_in[8];
    const float* Wn1  = (const float*)d_in[9];
    const float* bn1  = (const float*)d_in[10];
    const float* Wn2  = (const float*)d_in[11];
    const float* bn2  = (const float*)d_in[12];
    const float* Wd1  = (const float*)d_in[13];
    const float* bd1  = (const float*)d_in[14];
    const float* Wd2  = (const float*)d_in[15];
    const float* bd2  = (const float*)d_in[16];
    const int* s_fine = (const int*)d_in[17];
    const int* r_fine = (const int*)d_in[18];
    const int* s_ds   = (const int*)d_in[19];
    const int* r_ds   = (const int*)d_in[20];
    const int* s_p    = (const int*)d_in[21];
    const int* r_p    = (const int*)d_in[22];
    const int* s_us   = (const int*)d_in[23];
    const int* r_us   = (const int*)d_in[24];
    const int* uppool   = (const int*)d_in[25];
    const int* downpool = (const int*)d_in[26];

    const int N   = in_sizes[25];
    const int NP  = in_sizes[26];
    const int DIN = in_sizes[0] / N;   // 16
    const int E   = in_sizes[17];
    const int EDS = in_sizes[19];
    const int EP  = in_sizes[21];
    const int EUS = in_sizes[23];
    (void)DIN; (void)n_in; (void)out_size;

    const size_t bufElems = (size_t)N * DD;
    if (ws_size < 6 * bufElems * sizeof(float)) return;
    float* B0 = (float*)d_ws;             // h
    float* B1 = B0 + bufElems;            // msg (internal)
    float* B2 = B1 + bufElems;            // aggr (internal)
    float* B3 = B2 + bufElems;            // skip
    float* B4 = B3 + bufElems;            // tmp
    float* B5 = B4 + bufElems;            // tmp2

    auto blocks64 = [](int R) { return (R + 63) / 64; };
    auto row16    = [](int R) { return (int)(((long long)R * 16 + 255) / 256); };

    // One message-passing layer (message MLP hoisted to nodes + scatter-add).
    auto mp = [&](int l, const float* hin, const int* s, const int* r,
                  int n, int Eg, float* hout, const float* resid) {
        mlp2_wmma_kernel<64, false><<<blocks64(n), 128, 0, stream>>>(
            hin, nullptr,
            Wm1 + (size_t)l * 64 * 64, bm1 + (size_t)l * 64,
            Wm2 + (size_t)l * 64 * 64, bm2 + (size_t)l * 64,
            nullptr, B1, n);
        long long ztot = (long long)n * DD;
        zero_kernel<<<(int)((ztot + 255) / 256), 256, 0, stream>>>(B2, ztot);
        scatter_add_kernel<<<(int)(((long long)Eg * 16 + 255) / 256), 256, 0, stream>>>(
            B1, s, r, B2, Eg);
        mlp2_wmma_kernel<128, true><<<blocks64(n), 128, 0, stream>>>(
            hin, B2,
            Wn1 + (size_t)l * 128 * 64, bn1 + (size_t)l * 64,
            Wn2 + (size_t)l * 64 * 64,  bn2 + (size_t)l * 64,
            resid, hout, n);
    };

    // encode: h = MLP2(x)  (K1 = 16)
    mlp2_wmma_kernel<16, false><<<blocks64(N), 128, 0, stream>>>(
        x, nullptr, We1, be1, We2, be2, nullptr, B0, N);

    mp(0, B0, s_fine, r_fine, N, E, B4, nullptr);      // h -> B4
    mp(1, B4, s_fine, r_fine, N, E, B3, nullptr);      // h -> B3 (= skip)

    // downsample path
    gather_kernel<<<row16(N), 256, 0, stream>>>(B3, uppool, B0, N);   // h[uppool]
    mp(2, B0, s_ds, r_ds, N, EDS, B4, nullptr);                       // g -> B4
    gather_kernel<<<row16(NP), 256, 0, stream>>>(B4, downpool, B5, NP); // hp -> B5

    // bottleneck
    mp(3, B5, s_p, r_p, NP, EP, B4, nullptr);          // hp -> B4
    mp(4, B4, s_p, r_p, NP, EP, B5, nullptr);          // hp -> B5

    // upsample path
    gather_kernel<<<row16(NP), 256, 0, stream>>>(B5, downpool, B4, NP); // hp[downpool]
    mp(5, B4, s_us, r_us, NP, EUS, B5, nullptr);                        // g_p -> B5
    gather_kernel<<<row16(N), 256, 0, stream>>>(B5, uppool, B0, N);     // h -> B0

    // decode with skip connections (residual fused into node-MLP epilogue)
    mp(6, B0, s_fine, r_fine, N, E, B4, B3);           // h -> B4 (+skip)
    mp(7, B4, s_fine, r_fine, N, E, B0, B3);           // h -> B0 (+skip)

    decode_kernel<<<(N + 127) / 128, 128, 0, stream>>>(
        B0, Wd1, bd1, Wd2, bd2, (float*)d_out, N);
}
